// Attention_45578192945380
// MI455X (gfx1250) — matrix operand order repaired and checked by reading, not compiled, not run on hardware
//
#include <hip/hip_runtime.h>

typedef __attribute__((ext_vector_type(16))) __bf16 v16bf;
typedef __attribute__((ext_vector_type(8)))  __bf16 v8bf;
typedef __attribute__((ext_vector_type(8)))  float  v8f;
typedef __attribute__((ext_vector_type(4)))  unsigned int u32x4;
typedef __attribute__((ext_vector_type(8)))  int i32x8;
typedef __attribute__((ext_vector_type(4)))  int i32x4;

#define HEADS  8
#define DHEAD  64
#define DIM    512
#define NSEQ   2048
#define BATCH  4
#define NQKV   1536
#define ROWS   (BATCH * NSEQ)   // 8192

#if defined(__has_builtin)
#if __has_builtin(__builtin_amdgcn_tensor_load_to_lds)
#define HAVE_TDM 1
#endif
#if __has_builtin(__builtin_amdgcn_s_wait_tensorcnt)
#define HAVE_WTC 1
#endif
#endif

// ---------- helpers ----------

__device__ __forceinline__ unsigned short f2bf(float f) {
  union { float f; unsigned int u; } v; v.f = f;
  unsigned int r = v.u + 0x7FFFu + ((v.u >> 16) & 1u);   // round-to-nearest-even
  return (unsigned short)(r >> 16);
}

__device__ __forceinline__ v8f zero8() {
  v8f z;
#pragma unroll
  for (int i = 0; i < 8; ++i) z[i] = 0.0f;
  return z;
}

__device__ __forceinline__ v8f wmma_bf16(v16bf a, v16bf b, v8f c) {
  return __builtin_amdgcn_wmma_f32_16x16x32_bf16(
      /*neg_a=*/false, a, /*neg_b=*/false, b,
      /*c_mod=*/(short)0, c, /*reuse_a=*/false, /*reuse_b=*/false);
}

// A-fragment (16x32 bf16, M x K) from row-major matrix, per ISA A layout.
__device__ __forceinline__ v16bf load_a_rm(const unsigned short* A, int lda,
                                           int row0, int k0) {
  int lane = threadIdx.x & 31;
  int hi = lane >> 4;
  const unsigned short* p = A + (size_t)(row0 + (lane & 15)) * lda + k0;
  v8bf r0 = *(const v8bf*)(p + hi * 8);
  v8bf r1 = *(const v8bf*)(p + 16 + hi * 8);
  return __builtin_shufflevector(r0, r1, 0,1,2,3,4,5,6,7,8,9,10,11,12,13,14,15);
}

// B-fragment (32x16 bf16, K x N) from an N-major ("Bt") matrix: Bt[c*ldb + k].
__device__ __forceinline__ v16bf load_b_nm(const unsigned short* Bt, int ldb,
                                           int c0, int k0) {
  int lane = threadIdx.x & 31;
  const unsigned short* p =
      Bt + (size_t)(c0 + (lane & 15)) * ldb + k0 + ((lane >> 4) << 3);
  v8bf r0 = *(const v8bf*)(p);
  v8bf r1 = *(const v8bf*)(p + 16);
  return __builtin_shufflevector(r0, r1, 0,1,2,3,4,5,6,7,8,9,10,11,12,13,14,15);
}

#if HAVE_TDM
__device__ __forceinline__ void wait_tensorcnt0() {
#if HAVE_WTC
  __builtin_amdgcn_s_wait_tensorcnt((short)0);
#else
  asm volatile("s_wait_tensorcnt 0x0" ::: "memory");
#endif
}
__device__ __forceinline__ void wait_tensorcnt2() {
#if HAVE_WTC
  __builtin_amdgcn_s_wait_tensorcnt((short)2);
#else
  asm volatile("s_wait_tensorcnt 0x2" ::: "memory");
#endif
}

// TDM 2D tile load: bf16 elements (data_size code 1 = 2 bytes).
// Descriptor per CDNA5 ISA 8.3/8.4. Higher-dim groups zero (tile_dim2/3/4 = 0 -> unused).
__device__ __forceinline__ void tdm_load_2d(unsigned lds_addr, const void* gptr,
                                            unsigned tensor_d0, unsigned tensor_d1,
                                            unsigned tile_d0, unsigned tile_d1,
                                            unsigned stride0) {
  unsigned long long ga = (unsigned long long)gptr;
  u32x4 g0;
  g0[0] = 1u;                                             // count=1, user descriptor
  g0[1] = lds_addr;                                       // lds_addr (bytes)
  g0[2] = (unsigned)ga;                                   // global_addr[31:0]
  g0[3] = (unsigned)((ga >> 32) & 0x01FFFFFFu) | (2u << 30); // addr[56:32] | type=2
  i32x8 g1;
  g1[0] = (int)(1u << 16);                                // data_size=1 (2B), mask=0
  g1[1] = (int)((tensor_d0 & 0xFFFFu) << 16);             // tensor_dim0[15:0]
  g1[2] = (int)((tensor_d0 >> 16) | ((tensor_d1 & 0xFFFFu) << 16));
  g1[3] = (int)((tensor_d1 >> 16) | (tile_d0 << 16));     // tile_dim0
  g1[4] = (int)tile_d1;                                   // tile_dim1 (tile_dim2=0)
  g1[5] = (int)stride0;                                   // tensor_dim0_stride[31:0]
  g1[6] = 0;
  g1[7] = 0;
  i32x4 z4;
  z4[0] = 0; z4[1] = 0; z4[2] = 0; z4[3] = 0;
  i32x8 z8;
#pragma unroll
  for (int i = 0; i < 8; ++i) z8[i] = 0;
  __builtin_amdgcn_tensor_load_to_lds(g0, g1, z4, z4, z8, 0);
}
#endif

__device__ __forceinline__ unsigned lds_offset_of(const void* p) {
  // generic LDS pointers are {aperture_hi, lds_addr[31:0]} per the ISA aperture map
  return (unsigned)(unsigned long long)p;
}

// ---------- kernel 1: weight transpose + bf16 convert ----------

__global__ __launch_bounds__(256) void prep_w(const float* __restrict__ wqkv,
                                              const float* __restrict__ wout,
                                              unsigned short* __restrict__ WQ,
                                              unsigned short* __restrict__ WO) {
  int idx = blockIdx.x * 256 + threadIdx.x;
  if (idx < DIM * NQKV) {
    int k = idx / NQKV, c = idx % NQKV;
    WQ[(size_t)c * DIM + k] = f2bf(wqkv[idx]);
  }
  if (idx < DIM * DIM) {
    int k = idx / DIM, c = idx % DIM;
    WO[(size_t)c * DIM + k] = f2bf(wout[idx]);
  }
}

// ---------- kernel 2: LayerNorm -> bf16 ----------

__global__ __launch_bounds__(256) void ln_kernel(const float* __restrict__ x,
                                                 const float* __restrict__ gamma,
                                                 const float* __restrict__ betaln,
                                                 unsigned short* __restrict__ XN) {
  int wid = threadIdx.x >> 5, lane = threadIdx.x & 31;
  int row = blockIdx.x * 8 + wid;
  const float* xr = x + (size_t)row * DIM;
  float v[16];
  float s = 0.f;
#pragma unroll
  for (int i = 0; i < 16; ++i) { v[i] = xr[lane + 32 * i]; s += v[i]; }
#pragma unroll
  for (int m = 1; m < 32; m <<= 1) s += __shfl_xor(s, m, 32);
  float mu = s * (1.0f / DIM);
  float q = 0.f;
#pragma unroll
  for (int i = 0; i < 16; ++i) { float d = v[i] - mu; q += d * d; }
#pragma unroll
  for (int m = 1; m < 32; m <<= 1) q += __shfl_xor(q, m, 32);
  float rstd = rsqrtf(q * (1.0f / DIM) + 1e-5f);
#pragma unroll
  for (int i = 0; i < 16; ++i) {
    int c = lane + 32 * i;
    XN[(size_t)row * DIM + c] = f2bf((v[i] - mu) * rstd * gamma[c] + betaln[c]);
  }
}

// ---------- kernel 3: QKV GEMM (XN[8192,512] @ WqkvT) with per-head scatter ----------

__global__ __launch_bounds__(256) void qkv_gemm(const unsigned short* __restrict__ XN,
                                                const unsigned short* __restrict__ WQ,
                                                unsigned short* __restrict__ Qb,
                                                unsigned short* __restrict__ Kb,
                                                unsigned short* __restrict__ VT) {
  int wid = threadIdx.x >> 5, lane = threadIdx.x & 31;
  int g = blockIdx.x * 8 + wid;          // 12288 waves
  int mt = g / 24, nt = g % 24;
  int row0 = mt * 16, c0 = nt * 64;
  v8f acc[4];
#pragma unroll
  for (int ch = 0; ch < 4; ++ch) acc[ch] = zero8();
  for (int k0 = 0; k0 < DIM; k0 += 32) {
    v16bf a = load_a_rm(XN, DIM, row0, k0);
#pragma unroll
    for (int ch = 0; ch < 4; ++ch) {
      v16bf b = load_b_nm(WQ, DIM, c0 + ch * 16, k0);
      acc[ch] = wmma_bf16(a, b, acc[ch]);
    }
  }
  int nloc = lane & 15, mbase = (lane >> 4) * 8;
#pragma unroll
  for (int ch = 0; ch < 4; ++ch) {
#pragma unroll
    for (int r = 0; r < 8; ++r) {
      int c = c0 + ch * 16 + nloc;
      int row = row0 + mbase + r;
      int b = row >> 11, i = row & 2047;
      int which = c / 512, cc = c % 512;
      int h = cc >> 6, d = cc & 63;
      unsigned short val = f2bf(acc[ch][r]);
      size_t bh = (size_t)(b * HEADS + h);
      if (which == 0)      Qb[(bh * NSEQ + i) * DHEAD + d] = val;
      else if (which == 1) Kb[(bh * NSEQ + i) * DHEAD + d] = val;
      else                 VT[(bh * DHEAD + d) * NSEQ + i] = val;
    }
  }
}

// ---------- kernel 4: flash attention ----------
// Block = 8 waves sharing one (b,h); K/V j-tiles staged in LDS by the TDM
// (double-buffered), each wave owns one 16-row Q tile.

__global__ __launch_bounds__(256) void attn_kernel(const unsigned short* __restrict__ Qb,
                                                   const unsigned short* __restrict__ Kb,
                                                   const unsigned short* __restrict__ VT,
                                                   const float* __restrict__ pose,
                                                   const float* __restrict__ beta_p,
                                                   unsigned short* __restrict__ O) {
  __shared__ alignas(64) unsigned short bufK[2][32 * 64];  // [jloc][d]
  __shared__ alignas(64) unsigned short bufV[2][64 * 32];  // [d][jloc]
  __shared__ alignas(64) unsigned short lds_p[8 * 16 * 32];
  int wid = threadIdx.x >> 5, lane = threadIdx.x & 31;
  int bh = blockIdx.x >> 4;             // 0..31
  int grp = blockIdx.x & 15;
  int b = bh >> 3, h = bh & 7;
  int mt = grp * 8 + wid;               // 0..127
  const unsigned short* Qh = Qb + (size_t)bh * NSEQ * DHEAD;
  const unsigned short* Kh = Kb + (size_t)bh * NSEQ * DHEAD;
  const unsigned short* Vh = VT + (size_t)bh * DHEAD * NSEQ;
  int i0 = mt * 16;
  float beta = beta_p[0];
  const float scale = 0.125f;           // 1/sqrt(64)
  int nloc = lane & 15, mbase = (lane >> 4) * 8;

  v16bf qa0 = load_a_rm(Qh, DHEAD, i0, 0);
  v16bf qa1 = load_a_rm(Qh, DHEAD, i0, 32);

  v8f pbi;
#pragma unroll
  for (int r = 0; r < 8; ++r) pbi[r] = pose[b * NSEQ + i0 + mbase + r];

  v8f mprev, lsum, acc0 = zero8(), acc1 = zero8(), acc2 = zero8(), acc3 = zero8();
#pragma unroll
  for (int r = 0; r < 8; ++r) { mprev[r] = -3.0e38f; lsum[r] = 0.f; }

  unsigned short* pl = lds_p + wid * (16 * 32);

#if HAVE_TDM
  // prologue: DMA tile 0 into buffer 0 (K: 32 rows x 64, V: 64 rows x 32)
  if (wid == 0) {
    tdm_load_2d(lds_offset_of(&bufK[0][0]), Kh, DHEAD, NSEQ, DHEAD, 32, DHEAD);
    tdm_load_2d(lds_offset_of(&bufV[0][0]), Vh, NSEQ, DHEAD, 32, DHEAD, NSEQ);
  }
#endif

  for (int it = 0; it < NSEQ / 32; ++it) {
    int j0 = it * 32;
    int cur = it & 1;
#if HAVE_TDM
    if (wid == 0) {
      if (it + 1 < NSEQ / 32) {
        int jn = j0 + 32;
        tdm_load_2d(lds_offset_of(&bufK[cur ^ 1][0]),
                    Kh + (size_t)jn * DHEAD, DHEAD, NSEQ, DHEAD, 32, DHEAD);
        tdm_load_2d(lds_offset_of(&bufV[cur ^ 1][0]),
                    Vh + jn, NSEQ, DHEAD, 32, DHEAD, NSEQ);
        wait_tensorcnt2();   // current tile's pair complete (in-order), next in flight
      } else {
        wait_tensorcnt0();
      }
    }
#else
    // fallback: cooperative vector copy (256 threads x 16B per tile)
    {
      int t = threadIdx.x;
      *(v8bf*)(&bufK[cur][0] + t * 8) =
          *(const v8bf*)(Kh + (size_t)(j0 + (t >> 3)) * DHEAD + (t & 7) * 8);
      *(v8bf*)(&bufV[cur][0] + t * 8) =
          *(const v8bf*)(Vh + (size_t)(t >> 2) * NSEQ + j0 + (t & 3) * 8);
    }
#endif
    __syncthreads();

    const unsigned short* Kl = &bufK[cur][0];
    const unsigned short* Vl = &bufV[cur][0];

    // ---- dots = Q Kt for 16x32 block of columns (from LDS) ----
    v16bf kb00 = load_b_nm(Kl, DHEAD, 0,  0);
    v16bf kb01 = load_b_nm(Kl, DHEAD, 0,  32);
    v16bf kb10 = load_b_nm(Kl, DHEAD, 16, 0);
    v16bf kb11 = load_b_nm(Kl, DHEAD, 16, 32);
    v8f c0v = zero8(), c1v = zero8();
    c0v = wmma_bf16(qa0, kb00, c0v);
    c0v = wmma_bf16(qa1, kb01, c0v);
    c1v = wmma_bf16(qa0, kb10, c1v);
    c1v = wmma_bf16(qa1, kb11, c1v);

    float pbj0 = beta * pose[b * NSEQ + j0 + nloc];
    float pbj1 = beta * pose[b * NSEQ + j0 + 16 + nloc];
#pragma unroll
    for (int r = 0; r < 8; ++r) {
      float base = beta * pbi[r];
      c0v[r] = c0v[r] * scale + base + pbj0;
      c1v[r] = c1v[r] * scale + base + pbj1;
    }

    // ---- online softmax over 32 columns ----
    v8f mnewv;
#pragma unroll
    for (int r = 0; r < 8; ++r) {
      float m = fmaxf(c0v[r], c1v[r]);
      m = fmaxf(m, __shfl_xor(m, 1, 32));
      m = fmaxf(m, __shfl_xor(m, 2, 32));
      m = fmaxf(m, __shfl_xor(m, 4, 32));
      m = fmaxf(m, __shfl_xor(m, 8, 32));
      mnewv[r] = fmaxf(m, mprev[r]);
    }
#pragma unroll
    for (int r = 0; r < 8; ++r) {
      float alpha = __expf(mprev[r] - mnewv[r]);
      float p0 = __expf(c0v[r] - mnewv[r]);
      float p1 = __expf(c1v[r] - mnewv[r]);
      float rs = p0 + p1;
      rs += __shfl_xor(rs, 1, 32);
      rs += __shfl_xor(rs, 2, 32);
      rs += __shfl_xor(rs, 4, 32);
      rs += __shfl_xor(rs, 8, 32);
      lsum[r] = lsum[r] * alpha + rs;
      mprev[r] = mnewv[r];
      acc0[r] *= alpha; acc1[r] *= alpha; acc2[r] *= alpha; acc3[r] *= alpha;
      // relayout P (D-format -> row-major bf16) through per-wave LDS
      pl[(mbase + r) * 32 + nloc]      = f2bf(p0);
      pl[(mbase + r) * 32 + nloc + 16] = f2bf(p1);
    }
    asm volatile("s_wait_dscnt 0" ::: "memory");
    v16bf pa = load_a_rm(pl, 32, 0, 0);
    asm volatile("" ::: "memory");

    // ---- acc += P @ V (V tile in LDS, j-local K-dim) ----
    acc0 = wmma_bf16(pa, load_b_nm(Vl, 32, 0,  0), acc0);
    acc1 = wmma_bf16(pa, load_b_nm(Vl, 32, 16, 0), acc1);
    acc2 = wmma_bf16(pa, load_b_nm(Vl, 32, 32, 0), acc2);
    acc3 = wmma_bf16(pa, load_b_nm(Vl, 32, 48, 0), acc3);

    __syncthreads();
  }

  // ---- finalize and store bf16 O[b*n + i, h*64 + d] ----
  v8f inv;
#pragma unroll
  for (int r = 0; r < 8; ++r) inv[r] = 1.0f / lsum[r];
#pragma unroll
  for (int r = 0; r < 8; ++r) {
    size_t row = (size_t)(b * NSEQ + i0 + mbase + r);
    size_t col0 = (size_t)(h * DHEAD) + nloc;
    O[row * DIM + col0]      = f2bf(acc0[r] * inv[r]);
    O[row * DIM + col0 + 16] = f2bf(acc1[r] * inv[r]);
    O[row * DIM + col0 + 32] = f2bf(acc2[r] * inv[r]);
    O[row * DIM + col0 + 48] = f2bf(acc3[r] * inv[r]);
  }
}

// ---------- kernel 5: out projection (O[8192,512] @ WoutT) -> f32 ----------

__global__ __launch_bounds__(256) void outproj_gemm(const unsigned short* __restrict__ O,
                                                    const unsigned short* __restrict__ WO,
                                                    float* __restrict__ out) {
  int wid = threadIdx.x >> 5, lane = threadIdx.x & 31;
  int g = blockIdx.x * 8 + wid;          // 4096 waves
  int mt = g >> 3, nt = g & 7;
  int row0 = mt * 16, c0 = nt * 64;
  v8f acc[4];
#pragma unroll
  for (int ch = 0; ch < 4; ++ch) acc[ch] = zero8();
  for (int k0 = 0; k0 < DIM; k0 += 32) {
    v16bf a = load_a_rm(O, DIM, row0, k0);
#pragma unroll
    for (int ch = 0; ch < 4; ++ch) {
      v16bf bfr = load_b_nm(WO, DIM, c0 + ch * 16, k0);
      acc[ch] = wmma_bf16(a, bfr, acc[ch]);
    }
  }
  int nloc = lane & 15, mbase = (lane >> 4) * 8;
#pragma unroll
  for (int ch = 0; ch < 4; ++ch)
#pragma unroll
    for (int r = 0; r < 8; ++r)
      out[(size_t)(row0 + mbase + r) * DIM + c0 + ch * 16 + nloc] = acc[ch][r];
}

// ---------- host launcher ----------

extern "C" void kernel_launch(void* const* d_in, const int* in_sizes, int n_in,
                              void* d_out, int out_size, void* d_ws, size_t ws_size,
                              hipStream_t stream) {
  const float* x      = (const float*)d_in[0];
  const float* pose   = (const float*)d_in[1];
  const float* gamma  = (const float*)d_in[2];
  const float* betaln = (const float*)d_in[3];
  const float* wqkv   = (const float*)d_in[4];
  const float* wout   = (const float*)d_in[5];
  const float* beta   = (const float*)d_in[6];
  float* out = (float*)d_out;

  char* ws = (char*)d_ws;
  size_t off = 0;
  auto alloc = [&](size_t bytes) -> void* {
    void* p = ws + off;
    off = (off + bytes + 255) & ~(size_t)255;
    return p;
  };
  unsigned short* WQ = (unsigned short*)alloc((size_t)NQKV * DIM * 2);
  unsigned short* WO = (unsigned short*)alloc((size_t)DIM * DIM * 2);
  unsigned short* XN = (unsigned short*)alloc((size_t)ROWS * DIM * 2);
  unsigned short* Qb = (unsigned short*)alloc((size_t)BATCH * HEADS * NSEQ * DHEAD * 2);
  unsigned short* Kb = (unsigned short*)alloc((size_t)BATCH * HEADS * NSEQ * DHEAD * 2);
  unsigned short* VT = (unsigned short*)alloc((size_t)BATCH * HEADS * DHEAD * NSEQ * 2);
  unsigned short* O  = (unsigned short*)alloc((size_t)ROWS * DIM * 2);

  prep_w<<<(DIM * NQKV + 255) / 256, 256, 0, stream>>>(wqkv, wout, WQ, WO);
  ln_kernel<<<ROWS / 8, 256, 0, stream>>>(x, gamma, betaln, XN);
  qkv_gemm<<<(ROWS / 16) * (NQKV / 64) / 8, 256, 0, stream>>>(XN, WQ, Qb, Kb, VT);
  attn_kernel<<<BATCH * HEADS * 16, 256, 0, stream>>>(Qb, Kb, VT, pose, beta, O);
  outproj_gemm<<<(ROWS / 16) * (DIM / 64) / 8, 256, 0, stream>>>(O, WO, out);
}
